// Feature_11733850652929
// MI455X (gfx1250) — matrix operand order repaired and checked by reading, not compiled, not run on hardware
//
#include <hip/hip_runtime.h>
#include <cstdint>

typedef __attribute__((ext_vector_type(16))) _Float16 v16h;
typedef __attribute__((ext_vector_type(8)))  float    v8f;

#define NB    64      // batch
#define NA    128     // atoms
#define NK    127     // neighbors per atom
#define NTYP2 16      // type-pair count
#define D0    25
#define D1    50
#define D2    100
#define PAIRS (NA*NK) // 16256

union FragH { v16h h; unsigned u[8]; };

__device__ __forceinline__ unsigned short f16_bits(float f) {
  union { _Float16 h; unsigned short u; } cv;
  cv.h = (_Float16)f;
  return cv.u;
}

// gfx1250 has a hardware transcendental v_tanh_f32 (ISA lists TANH among
// TRANS32 ops, which co-execute with XDL WMMA).  Use it when the toolchain
// exposes the builtin; otherwise fall back to the branch-free
// 1 - 2/(exp(2x)+1) sequence (v_exp_f32 + v_rcp_f32, saturates correctly).
__device__ __forceinline__ float fast_tanh(float x) {
#if __has_builtin(__builtin_amdgcn_tanhf)
  return __builtin_amdgcn_tanhf(x);
#else
  float e = __expf(2.0f * x);
  return 1.0f - 2.0f * __builtin_amdgcn_rcpf(e + 1.0f);
#endif
}

// A-fragment (16x32 f16) element map:  value (row M, col K) lives at
//   lane = M + 16*((K>>3)&1),  halfword-elem = (K&16 ? 8 : 0) + (K&7)
// (matches ISA 7.12.2 16-bit A table; elem e of v16h = VGPR e/2, half e%2).

// ---------------------------------------------------------------------------
// Kernel 1: local environment  r = 1/d,  a = rij / d^2
// ---------------------------------------------------------------------------
__global__ void env_kernel(const float* __restrict__ coords,
                           float* __restrict__ r_ws,
                           float* __restrict__ a_ws) {
  int bn = blockIdx.x;              // b*128 + n
  int b  = bn >> 7, n = bn & 127;
  int k  = threadIdx.x;
  if (k >= NK) return;
  int nb = (k < n) ? k : k + 1;
  const float* ci = coords + (size_t)bn * 3;
  const float* cj = coords + ((size_t)(b * NA + nb)) * 3;
  float dx = ci[0] - cj[0], dy = ci[1] - cj[1], dz = ci[2] - cj[2];
  float d2  = dx*dx + dy*dy + dz*dz;
  float inv = 1.0f / sqrtf(d2);
  float inv2 = inv * inv;
  size_t p = (size_t)bn * NK + k;
  r_ws[p]       = inv;
  a_ws[p*3 + 0] = dx * inv2;
  a_ws[p*3 + 1] = dy * inv2;
  a_ws[p*3 + 2] = dz * inv2;
}

// ---------------------------------------------------------------------------
// Kernel 2: pre-pack W1/W2 into WMMA B-fragments, per-lane CONTIGUOUS:
// layout [tile][lane][v] so each lane fetches its 8 dwords with 2x b128.
// B (KxN) f16 layout (table order): lanes 0-15 hold K=0..7,16..23,
// lanes 16-31 hold K=8..15,24..31; N = lane%16.  B[k][n] = W[n][k], zero-padded.
// ---------------------------------------------------------------------------
__global__ void pack_w_kernel(const float* __restrict__ W1,
                              const float* __restrict__ W2,
                              unsigned* __restrict__ w1f,
                              unsigned* __restrict__ w2f) {
  int idx = blockIdx.x;
  int tid = threadIdx.x;
  // layer1: 4 N-tiles x 32 lanes x 8 VGPR
  for (int e = tid; e < 4*32*8; e += 256) {
    int v = e & 7, lane = (e >> 3) & 31, nt = e >> 8;
    int ncol = nt*16 + (lane & 15);
    int kb   = ((v < 4) ? 2*v : 16 + 2*(v-4)) + ((lane < 16) ? 0 : 8);
    unsigned pk = 0;
    for (int hh = 0; hh < 2; ++hh) {
      int kk = kb + hh;
      float val = (kk < D0 && ncol < D1)
                  ? W1[((size_t)idx*D1 + ncol)*D0 + kk] : 0.0f;
      pk |= (unsigned)f16_bits(val) << (16*hh);
    }
    w1f[(size_t)idx*1024 + e] = pk;
  }
  // layer2: 7 N-tiles x 2 K-chunks x 32 lanes x 8 VGPR
  for (int e = tid; e < 7*2*32*8; e += 256) {
    int v = e & 7, lane = (e >> 3) & 31, kc = (e >> 8) & 1, nt = e >> 9;
    int ncol = nt*16 + (lane & 15);
    int kb   = kc*32 + ((v < 4) ? 2*v : 16 + 2*(v-4)) + ((lane < 16) ? 0 : 8);
    unsigned pk = 0;
    for (int hh = 0; hh < 2; ++hh) {
      int kk = kb + hh;
      float val = (kk < D1 && ncol < D2)
                  ? W2[((size_t)idx*D2 + ncol)*D1 + kk] : 0.0f;
      pk |= (unsigned)f16_bits(val) << (16*hh);
    }
    w2f[(size_t)idx*3584 + e] = pk;
  }
}

// ---------------------------------------------------------------------------
// Kernel 3: typed MLP via WMMA.  One block per (n,k) pair; wave w handles
// batches [16w,16w+16) -> a 16-row tile shares a single idx / weight set.
// ---------------------------------------------------------------------------
__global__ void __launch_bounds__(128)
mlp_kernel(const float* __restrict__ r_ws,
           const int*   __restrict__ types,
           const float* __restrict__ W0, const float* __restrict__ b0v,
           const float* __restrict__ b1v, const float* __restrict__ b2v,
           const unsigned* __restrict__ w1f, const unsigned* __restrict__ w2f,
           _Float16* __restrict__ G) {
  __shared__ float y0s[4][16][32];                    // per-wave y0 (f32, row-major)
  __shared__ __align__(32) _Float16 y1f[4][32][32];   // per-wave y1 in A-frag order

  int p  = blockIdx.x;
  int n  = p / NK, k = p - n * NK;
  int nb = (k < n) ? k : k + 1;
  int idx = types[n] * 4 + types[nb];

  int tid   = threadIdx.x;
  int w     = tid >> 5, lane = tid & 31;
  int m     = lane & 15;         // local row (batch) / local col
  int hi    = lane >> 4;         // lane half
  int bbase = w * 16;

  float x = r_ws[((size_t)(bbase + m) * NA + n) * NK + k];

  const float* W0g = W0  + idx * D0;
  const float* b0g = b0v + idx * D0;

  // ---- layer 0 (scalar f32) directly into A-fragment registers
  FragH a1;
  #pragma unroll
  for (int v = 0; v < 8; ++v) {
    int kbase = ((v < 4) ? 2*v : 16 + 2*(v-4)) + hi*8;
    #pragma unroll
    for (int hh = 0; hh < 2; ++hh) {
      int kk = kbase + hh;
      float y = (kk < D0) ? fast_tanh(W0g[kk] * x + b0g[kk]) : 0.0f;
      a1.h[2*v + hh] = (_Float16)y;
      y0s[w][m][kk]  = y;
    }
  }
  __syncthreads();

  // ---- layer 1: y1 = tanh(y0*W1^T + b1) + [y0,y0]  (N=50 pad 64, K=25 pad 32)
  const float* b1g = b1v + idx * D1;
  const FragH* w1fp = (const FragH*)w1f;
  #pragma unroll
  for (int nt = 0; nt < 4; ++nt) {
    FragH bf = w1fp[(size_t)(idx*4 + nt)*32 + lane];   // 2x global_load_b128
    v8f c = {};
    c = __builtin_amdgcn_wmma_f32_16x16x32_f16(false, a1.h, false, bf.h,
                                               (short)0, c, false, false);
    int j = nt*16 + m;                    // output neuron (C layout: N = lane%16)
    // A-frag target coordinates for column j of the layer-2 A matrix
    int K      = j & 31;
    int elem   = (j >> 5)*16 + ((K & 16) ? 8 : 0) + (K & 7);
    int lshift = 16 * ((K >> 3) & 1);
    float bias = (j < D1) ? b1g[j] : 0.0f;
    #pragma unroll
    for (int r = 0; r < 8; ++r) {         // C layout: M = r + 8*hi
      int row = r + hi*8;
      float val = 0.0f;
      if (j < D1)
        val = fast_tanh(c[r] + bias) + y0s[w][row][j % D0];
      y1f[w][row + lshift][elem] = (_Float16)val;
    }
  }
  __syncthreads();

  // ---- A-fragments for layer 2: contiguous b128 LDS loads
  FragH a2lo = *(const FragH*)&y1f[w][lane][0];
  FragH a2hi = *(const FragH*)&y1f[w][lane][16];

  // ---- layer 2: G = tanh(y1*W2^T + b2) + [y1,y1] (N=100 pad 112, K=50 pad 64)
  const float* b2g = b2v + idx * D2;
  const FragH* w2fp = (const FragH*)w2f;
  #pragma unroll
  for (int nt = 0; nt < 7; ++nt) {
    FragH bf0 = w2fp[(size_t)((idx*7 + nt)*2 + 0)*32 + lane];
    FragH bf1 = w2fp[(size_t)((idx*7 + nt)*2 + 1)*32 + lane];
    v8f c = {};
    c = __builtin_amdgcn_wmma_f32_16x16x32_f16(false, a2lo.h, false, bf0.h,
                                               (short)0, c, false, false);
    c = __builtin_amdgcn_wmma_f32_16x16x32_f16(false, a2hi.h, false, bf1.h,
                                               (short)0, c, false, false);
    int mc = nt*16 + m;
    if (mc < D2) {
      // residual column coordinates in the y1 A-frag layout
      int cr     = mc % D1;
      int Kr     = cr & 31;
      int elemr  = (cr >> 5)*16 + ((Kr & 16) ? 8 : 0) + (Kr & 7);
      int lshr   = 16 * ((Kr >> 3) & 1);
      float bias = b2g[mc];
      #pragma unroll
      for (int r = 0; r < 8; ++r) {
        int row = r + hi*8;
        float val = fast_tanh(c[r] + bias) + (float)y1f[w][row + lshr][elemr];
        G[(((size_t)(bbase + row) * NA + n) * NK + k) * D2 + mc] = (_Float16)val;
      }
    }
  }
}

// ---------------------------------------------------------------------------
// Kernel 4: geometric contraction per (b,n):
//   t[c][a] = sum_k a[k][c] G[k][a] ; s[k][a] = sum_c a[k][c] t[c][a]
//   out[m][a] = sum_k G[k][m] s[k][a]
// ---------------------------------------------------------------------------
__global__ void __launch_bounds__(128)
geom_kernel(const float* __restrict__ a_ws, const _Float16* __restrict__ G,
            float* __restrict__ out) {
  __shared__ float Gs[NK * D2];                  // 50.8 KB
  __shared__ float as[NK][3];
  __shared__ float tm[12];                       // t[c*4+a]
  __shared__ __align__(16) float sm[NK][4];
  int bn  = blockIdx.x;
  int tid = threadIdx.x;
  const unsigned* Gp = (const unsigned*)(G + (size_t)bn * NK * D2);
  for (int i = tid; i < (NK*D2)/2; i += 128) {   // packed dword loads
    union { unsigned u; _Float16 h[2]; } cv; cv.u = Gp[i];
    Gs[2*i]     = (float)cv.h[0];
    Gs[2*i + 1] = (float)cv.h[1];
  }
  const float* ap = a_ws + (size_t)bn * NK * 3;
  for (int i = tid; i < NK*3; i += 128) as[i / 3][i % 3] = ap[i];
  __syncthreads();
  if (tid < 12) {
    int c = tid >> 2, a = tid & 3;
    float s = 0.0f;
    for (int kk = 0; kk < NK; ++kk) s += as[kk][c] * Gs[kk*D2 + a];
    tm[tid] = s;
  }
  __syncthreads();
  if (tid < NK) {
    float c0 = as[tid][0], c1 = as[tid][1], c2 = as[tid][2];
    float4 sv;
    sv.x = c0*tm[0] + c1*tm[4] + c2*tm[8];
    sv.y = c0*tm[1] + c1*tm[5] + c2*tm[9];
    sv.z = c0*tm[2] + c1*tm[6] + c2*tm[10];
    sv.w = c0*tm[3] + c1*tm[7] + c2*tm[11];
    *(float4*)&sm[tid][0] = sv;
  }
  __syncthreads();
  if (tid < D2) {                                // one neuron per thread
    float4 acc = make_float4(0.0f, 0.0f, 0.0f, 0.0f);
    for (int kk = 0; kk < NK; ++kk) {
      float g   = Gs[kk*D2 + tid];
      float4 sv = *(const float4*)&sm[kk][0];
      acc.x += g*sv.x; acc.y += g*sv.y; acc.z += g*sv.z; acc.w += g*sv.w;
    }
    *(float4*)(out + (size_t)bn * (D2*4) + tid*4) = acc;   // b128 store
  }
}

// ---------------------------------------------------------------------------
extern "C" void kernel_launch(void* const* d_in, const int* in_sizes, int n_in,
                              void* d_out, int out_size, void* d_ws, size_t ws_size,
                              hipStream_t stream) {
  (void)in_sizes; (void)n_in; (void)out_size; (void)ws_size;
  const float* coords = (const float*)d_in[0];
  const int*   types  = (const int*)d_in[1];
  const float* W0 = (const float*)d_in[2];
  const float* b0 = (const float*)d_in[3];
  const float* W1 = (const float*)d_in[4];
  const float* b1 = (const float*)d_in[5];
  const float* W2 = (const float*)d_in[6];
  const float* b2 = (const float*)d_in[7];

  char* ws = (char*)d_ws;
  size_t off = 0;
  auto carve = [&](size_t bytes) -> char* {
    char* p = ws + off;
    off = (off + bytes + 255) & ~(size_t)255;
    return p;
  };
  float*    r_ws = (float*)   carve((size_t)NB * PAIRS * sizeof(float));
  float*    a_ws = (float*)   carve((size_t)NB * PAIRS * 3 * sizeof(float));
  unsigned* w1f  = (unsigned*)carve((size_t)NTYP2 * 1024 * sizeof(unsigned));
  unsigned* w2f  = (unsigned*)carve((size_t)NTYP2 * 3584 * sizeof(unsigned));
  _Float16* G    = (_Float16*)carve((size_t)NB * PAIRS * D2 * sizeof(_Float16));

  env_kernel   <<<NB * NA, 128, 0, stream>>>(coords, r_ws, a_ws);
  pack_w_kernel<<<NTYP2,   256, 0, stream>>>(W1, W2, w1f, w2f);
  mlp_kernel   <<<PAIRS,   128, 0, stream>>>(r_ws, types, W0, b0, b1, b2, w1f, w2f, G);
  geom_kernel  <<<NB * NA, 128, 0, stream>>>(a_ws, G, (float*)d_out);
}
